// WLN_Layer_970662609323
// MI455X (gfx1250) — matrix operand order repaired and checked by reading, not compiled, not run on hardware
//
#include <hip/hip_runtime.h>
#include <hip/hip_bf16.h>
#include <cstdint>

// ---------------------------------------------------------------------------
// WLN layer for MI455X (gfx1250, wave32, WMMA + TDM gather).
// - All four GEMMs run on v_wmma_f32_16x16x32_f16 (f32 accumulate).
// - Generic GEMM: 32x64 tile per wave (8 WMMAs / K-step, A reused 4x, B 2x).
// - Fused neighbor kernel: per-atom 16x320 A tile gathered into LDS by the
//   Tensor Data Mover in gather mode (row indices from atom_graph), then two
//   WMMAs per K-step against W_nei_atom / W_U2; slot-reduction via shfl_xor.
// ---------------------------------------------------------------------------

typedef __attribute__((ext_vector_type(16))) _Float16 v16h;
typedef __attribute__((ext_vector_type(8)))  _Float16 v8h;
typedef __attribute__((ext_vector_type(8)))  float    v8f;
typedef __attribute__((ext_vector_type(4)))  uint32_t v4u;
typedef __attribute__((ext_vector_type(8)))  uint32_t v8u;

#define BB   128
#define NN   150
#define NBND 300
#define KNB  10
#define HH   300
#define FAA  83
#define FBB  6
#define DEPTH 3

#define MM    (BB * NN)     // 19200 rows (multiple of 32)
#define HP    320           // H padded for WMMA K-loops AND output tiles
#define NT    20            // output column tiles (320/16)
#define NG4   5             // groups of 4 column tiles
#define FAP   96            // FA padded to 32
#define NLP   640           // 2H padded to 32

// ---- WMMA operand loaders --------------------------------------------------
__device__ __forceinline__ v16h load_a16(const _Float16* p) {
  v8h lo = *(const v8h*)(p);
  v8h hi = *(const v8h*)(p + 16);
  return __builtin_shufflevector(lo, hi, 0,1,2,3,4,5,6,7,8,9,10,11,12,13,14,15);
}
__device__ __forceinline__ v16h load_b16(const _Float16* p) {
  v8h lo = *(const v8h*)(p);
  v8h hi = *(const v8h*)(p + 8);
  return __builtin_shufflevector(lo, hi, 0,1,2,3,4,5,6,7,8,9,10,11,12,13,14,15);
}
__device__ __forceinline__ v8f wmma_f16(v16h a, v16h b, v8f c) {
  return __builtin_amdgcn_wmma_f32_16x16x32_f16(false, a, false, b, (short)0, c,
                                                false, false);
}

// ---- utility kernels -------------------------------------------------------
__global__ void zero_half(_Float16* p, long n) {
  long i = (long)blockIdx.x * blockDim.x + threadIdx.x;
  if (i < n) p[i] = (_Float16)0.f;
}

__global__ void cvt_input_atom(const float* __restrict__ ia, _Float16* __restrict__ o) {
  long i = (long)blockIdx.x * blockDim.x + threadIdx.x;
  if (i >= (long)MM * FAP) return;
  long r = i / FAP, c = i - r * FAP;
  o[i] = (c < FAA) ? (_Float16)ia[r * FAA + c] : (_Float16)0.f;
}

// Pack W f32 [K x Ncols] into WMMA-B register layout:
// out[((kt*NT + nt)*32 + lane)*16 + j]; col = nt*16+(lane&15),
// elem j <-> K = kt*32 + (lane>>4)*8 + j + (j>=8 ? 8 : 0). Zero outside [Kdim, Ncols].
__global__ void pack_b_tiles(const float* __restrict__ W, _Float16* __restrict__ out,
                             int Kdim, int Ncols, int Ktiles) {
  int g = blockIdx.x * blockDim.x + threadIdx.x;
  if (g >= Ktiles * NT * 32) return;
  int lane = g & 31;
  int t = g >> 5;
  int nt = t % NT;
  int kt = t / NT;
  int col = nt * 16 + (lane & 15);
  int kbase = kt * 32 + (lane >> 4) * 8;
  _Float16* dst = out + (long)g * 16;
#pragma unroll
  for (int j = 0; j < 16; ++j) {
    int k = kbase + j + (j >= 8 ? 8 : 0);
    dst[j] = (k < Kdim && col < Ncols) ? (_Float16)W[(long)k * Ncols + col]
                                       : (_Float16)0.f;
  }
}

// ---- generic WMMA GEMM: 32x64 tile per wave --------------------------------
__global__ __launch_bounds__(128) void gemm32x64_wmma(
    const _Float16* __restrict__ A, const _Float16* __restrict__ Bp,
    const float* __restrict__ bias, float* __restrict__ C,
    _Float16* __restrict__ C16, int MG, int Ktiles, int Kp,
    int Np, int Np16, int do_relu) {
  int wave = blockIdx.x * (blockDim.x >> 5) + (threadIdx.x >> 5);
  int lane = threadIdx.x & 31;
  if (wave >= MG * NG4) return;
  int mg = wave / NG4, ng = wave % NG4;
  int hi8 = (lane >> 4) * 8;
  const _Float16* arow0 = A + (long)(mg * 32 + (lane & 15)) * Kp + hi8;
  const _Float16* arow1 = arow0 + (long)16 * Kp;
  const _Float16* bptr  = Bp + ((long)(ng * 4) * 32 + lane) * 16;
  const long bstep = (long)NT * 512;          // halves per K-tile of packed B
  v8f acc[2][4] = {};
  for (int kt = 0; kt < Ktiles; ++kt) {
    v16h a0 = load_a16(arow0 + kt * 32);
    v16h a1 = load_a16(arow1 + kt * 32);
#pragma unroll
    for (int u = 0; u < 4; ++u) {
      v16h b = load_b16(bptr + kt * bstep + u * 512);
      acc[0][u] = wmma_f16(a0, b, acc[0][u]);
      acc[1][u] = wmma_f16(a1, b, acc[1][u]);
    }
  }
  int colb = ng * 64 + (lane & 15);
  int mbase = mg * 32 + (lane >> 4) * 8;
#pragma unroll
  for (int u = 0; u < 4; ++u) {
    int col = colb + u * 16;
    float bv = (bias != nullptr && col < HH) ? bias[col] : 0.f;
#pragma unroll
    for (int h = 0; h < 2; ++h) {
#pragma unroll
      for (int i = 0; i < 8; ++i) {
        float v = acc[h][u][i] + bv;
        if (do_relu) v = v > 0.f ? v : 0.f;
        long r = mbase + h * 16 + i;
        C[r * Np + col] = v;
        if (C16 != nullptr) C16[r * Np16 + col] = (_Float16)v;
      }
    }
  }
}

// ---- fused per-atom neighbor kernel with TDM gather ------------------------
__global__ __launch_bounds__(256) void wln_neighbor_wmma(
    const _Float16* __restrict__ af16, const float* __restrict__ input_bond,
    const int* __restrict__ atom_graph, const int* __restrict__ bond_graph,
    const int* __restrict__ num_nbs, const _Float16* __restrict__ pWna,
    const _Float16* __restrict__ pWU2, const float* __restrict__ Wnb,
    const float* __restrict__ bU2, float* __restrict__ fnei,
    float* __restrict__ nlab) {
  __shared__ __align__(16) _Float16 As[16 * HP];
  __shared__ float fb[16][8];
  __shared__ int gRow[16], gBond[16];
  __shared__ int nnb_s;

  const int atom = blockIdx.x;
  const int tid = threadIdx.x;
  if (tid < 16) {
    int r = -1, bd = -1;
    if (tid < KNB) {
      int base = (atom * KNB + tid) * 2;
      r  = atom_graph[base] * NN   + atom_graph[base + 1];
      bd = bond_graph[base] * NBND + bond_graph[base + 1];
    }
    gRow[tid] = r;
    gBond[tid] = bd;
  }
  if (tid == 16) nnb_s = num_nbs[atom];

  // ---- TDM gather: 10 rows x 320 halfs of af16 -> As rows 0..9 (wave 0) ----
  if (tid < 32) {
    uint32_t idx[16];
#pragma unroll
    for (int j = 0; j < 16; ++j) idx[j] = 0u;
    for (int j = 0; j < KNB; ++j) {
      int base = (atom * KNB + j) * 2;
      int r = atom_graph[base] * NN + atom_graph[base + 1];
      idx[j] = (uint32_t)__builtin_amdgcn_readfirstlane(r);
    }
    uint64_t ga  = (uint64_t)(uintptr_t)af16;
    uint32_t lds = (uint32_t)(uintptr_t)(&As[0]);
    v4u g0;
    g0[0] = 0x80000001u;                       // count=1, gather_mode=1 (16-bit idx)
    g0[1] = lds;                               // lds_addr
    g0[2] = (uint32_t)ga;                      // global_addr[31:0]
    g0[3] = (uint32_t)((ga >> 32) & 0x01FFFFFFu) | (2u << 30);  // addr[56:32] | type=2
    v8u g1;
    g1[0] = (1u << 16);                        // data_size = 2 bytes
    g1[1] = ((uint32_t)HP & 0xFFFFu) << 16;    // tensor_dim0 = 320 (low 16)
    g1[2] = ((uint32_t)MM & 0xFFFFu) << 16;    // dim0 hi16=0 | tensor_dim1 = 19200
    g1[3] = ((uint32_t)MM >> 16) | (((uint32_t)HP & 0xFFFFu) << 16); // | tile_dim0=320
    g1[4] = (uint32_t)KNB;                     // tile_dim1 = 10 valid indices
    g1[5] = (uint32_t)HP;                      // tensor_dim0_stride = 320 (low 32)
    g1[6] = 0u;
    g1[7] = 0u;
    v4u g2, g3;
#pragma unroll
    for (int j = 0; j < 4; ++j) {
      g2[j] = idx[2 * j]     | (idx[2 * j + 1] << 16);
      g3[j] = idx[2 * j + 8] | (idx[2 * j + 9] << 16);
    }
    asm volatile("tensor_load_to_lds %0, %1, %2, %3"
                 :: "s"(g0), "s"(g1), "s"(g2), "s"(g3) : "memory");
    __builtin_amdgcn_s_wait_tensorcnt(0);
  }
  __syncthreads();

  // Remainder staging (after TDM; overwrites its cols 300..319): zero rows 10..15,
  // bond features into cols 300..305 of rows 0..9, zero cols 306..319.
  for (int i = tid; i < 16 * HP; i += 256) {
    int s = i / HP, c = i - s * HP;
    if (s >= KNB) {
      As[i] = (_Float16)0.f;
    } else if (c >= HH) {
      _Float16 v = (_Float16)0.f;
      if (c < HH + FBB && gBond[s] >= 0)
        v = (_Float16)input_bond[(long)gBond[s] * FBB + (c - HH)];
      As[i] = v;
    }
  }
  for (int i = tid; i < 16 * 8; i += 256) {
    int s = i >> 3, c = i & 7;
    float v = 0.f;
    if (c < FBB && gBond[s] >= 0) v = input_bond[(long)gBond[s] * FBB + c];
    fb[s][c] = v;
  }
  __syncthreads();

  const int lane = tid & 31;
  const int wv = tid >> 5;
  const int hi8 = (lane >> 4) * 8;
  const _Float16* arow = &As[(lane & 15) * HP + hi8];
  const int nnb = nnb_s;
  const long bstep = (long)NT * 512;

  for (int nt = wv; nt < NT; nt += 8) {        // wave-uniform; EXEC full for WMMA
    v8f acc1 = {};
    v8f acc2 = {};
    const _Float16* b1 = pWna + ((long)nt * 32 + lane) * 16;
    const _Float16* b2 = pWU2 + ((long)nt * 32 + lane) * 16;
    for (int kt = 0; kt < HP / 32; ++kt) {
      v16h a  = load_a16(arow + kt * 32);      // ds_load_b128 from LDS
      v16h w1 = load_b16(b1 + kt * bstep);
      v16h w2 = load_b16(b2 + kt * bstep);
      acc1 = wmma_f16(a, w1, acc1);
      acc2 = wmma_f16(a, w2, acc2);
    }
    int col = nt * 16 + (lane & 15);
    int mbase = (lane >> 4) * 8;
    float wcol[6];
#pragma unroll
    for (int j = 0; j < 6; ++j) wcol[j] = (col < HH) ? Wnb[j * HH + col] : 0.f;
    float bv = (col < HH) ? bU2[col] : 0.f;
    float s1 = 0.f, s2 = 0.f;
#pragma unroll
    for (int i = 0; i < 8; ++i) {
      int s = mbase + i;
      float m = (s < nnb) ? 1.f : 0.f;
      float hnb = fb[s][0] * wcol[0] + fb[s][1] * wcol[1] + fb[s][2] * wcol[2] +
                  fb[s][3] * wcol[3] + fb[s][4] * wcol[4] + fb[s][5] * wcol[5];
      s1 += acc1[i] * hnb * m;                 // h_nei = (A@Wna)*h_nei_bond
      float pl = acc2[i] + bv;                 // pre_label = relu(A@WU2 + b)
      pl = pl > 0.f ? pl : 0.f;
      s2 += pl * m;
    }
    s1 += __shfl_xor(s1, 16, 32);              // reduce slot halves (wave32)
    s2 += __shfl_xor(s2, 16, 32);
    if (lane < 16) {
      fnei[(long)atom * HP + col] = s1;
      nlab[(long)atom * HP + col] = s2;
    }
  }
}

__global__ void wln_kernels_out(const float* __restrict__ fnei,
                                const float* __restrict__ fself,
                                const float* __restrict__ nmask,
                                float* __restrict__ out) {
  long i = (long)blockIdx.x * blockDim.x + threadIdx.x;
  if (i >= (long)MM * HH) return;
  long r = i / HH, c = i - r * HH;
  out[i] = fnei[r * HP + c] * fself[r * HP + c] * nmask[r];
}

__global__ void wln_build_newlab(const _Float16* __restrict__ af16,
                                 const float* __restrict__ nlab,
                                 _Float16* __restrict__ nl16) {
  long i = (long)blockIdx.x * blockDim.x + threadIdx.x;
  if (i >= (long)MM * NLP) return;
  long r = i / NLP, c = i - r * NLP;
  _Float16 v = (_Float16)0.f;
  if (c < HH)          v = af16[r * HP + c];
  else if (c < 2 * HH) v = (_Float16)nlab[r * HP + (c - HH)];
  nl16[i] = v;
}

__global__ void wln_copy_af(const float* __restrict__ af32, float* __restrict__ out) {
  long i = (long)blockIdx.x * blockDim.x + threadIdx.x;
  if (i >= (long)MM * HH) return;
  long r = i / HH, c = i - r * HH;
  out[i] = af32[r * HP + c];
}

// ---------------------------------------------------------------------------
extern "C" void kernel_launch(void* const* d_in, const int* in_sizes, int n_in,
                              void* d_out, int out_size, void* d_ws, size_t ws_size,
                              hipStream_t stream) {
  (void)in_sizes; (void)n_in; (void)out_size; (void)ws_size;
  const float* input_atom = (const float*)d_in[0];
  const float* input_bond = (const float*)d_in[1];
  const int*   atom_graph = (const int*)d_in[2];
  const int*   bond_graph = (const int*)d_in[3];
  const int*   num_nbs    = (const int*)d_in[4];
  const float* node_mask  = (const float*)d_in[5];
  const float* W_atom     = (const float*)d_in[6];
  const float* W_nei_atom = (const float*)d_in[7];
  const float* W_nei_bond = (const float*)d_in[8];
  const float* W_self     = (const float*)d_in[9];
  const float* W_U2       = (const float*)d_in[10];
  const float* b_U2       = (const float*)d_in[11];
  const float* W_U1       = (const float*)d_in[12];
  const float* b_U1       = (const float*)d_in[13];
  float* out = (float*)d_out;

  char* p = (char*)d_ws;
  auto alloc = [&](size_t bytes) -> void* {
    void* r = (void*)p;
    p += (bytes + 255) & ~(size_t)255;
    return r;
  };
  _Float16* af16  = (_Float16*)alloc((size_t)MM * HP * 2);
  _Float16* ia16  = (_Float16*)alloc((size_t)MM * FAP * 2);
  _Float16* nl16  = (_Float16*)alloc((size_t)MM * NLP * 2);
  float* af32     = (float*)alloc((size_t)MM * HP * 4);
  float* fnei     = (float*)alloc((size_t)MM * HP * 4);
  float* fself    = (float*)alloc((size_t)MM * HP * 4);
  float* nlab     = (float*)alloc((size_t)MM * HP * 4);
  _Float16* pWatom = (_Float16*)alloc((size_t)3  * NT * 512 * 2);
  _Float16* pWna   = (_Float16*)alloc((size_t)10 * NT * 512 * 2);
  _Float16* pWU2   = (_Float16*)alloc((size_t)10 * NT * 512 * 2);
  _Float16* pWself = (_Float16*)alloc((size_t)10 * NT * 512 * 2);
  _Float16* pWU1   = (_Float16*)alloc((size_t)20 * NT * 512 * 2);

  const int TPB = 256;
  auto blk = [](long n, int t) { return (int)((n + t - 1) / t); };

  zero_half<<<blk((long)MM * HP, TPB), TPB, 0, stream>>>(af16, (long)MM * HP);
  cvt_input_atom<<<blk((long)MM * FAP, TPB), TPB, 0, stream>>>(input_atom, ia16);
  pack_b_tiles<<<blk(3  * NT * 32, TPB), TPB, 0, stream>>>(W_atom,     pWatom, FAA,      HH, 3);
  pack_b_tiles<<<blk(10 * NT * 32, TPB), TPB, 0, stream>>>(W_nei_atom, pWna,   HH,       HH, 10);
  pack_b_tiles<<<blk(10 * NT * 32, TPB), TPB, 0, stream>>>(W_U2,       pWU2,   HH + FBB, HH, 10);
  pack_b_tiles<<<blk(10 * NT * 32, TPB), TPB, 0, stream>>>(W_self,     pWself, HH,       HH, 10);
  pack_b_tiles<<<blk(20 * NT * 32, TPB), TPB, 0, stream>>>(W_U1,       pWU1,   2 * HH,   HH, 20);

  const int MG = MM / 32;                          // 600 row-groups
  const int gemmBlocks = blk((long)MG * NG4, 4);   // 4 waves / 128-thread block

  // atom_features = input_atom @ W_atom
  gemm32x64_wmma<<<gemmBlocks, 128, 0, stream>>>(ia16, pWatom, nullptr, af32, af16,
                                                 MG, 3, FAP, HP, HP, 0);

  for (int d = 0; d < DEPTH; ++d) {
    wln_neighbor_wmma<<<MM, 256, 0, stream>>>(af16, input_bond, atom_graph,
                                              bond_graph, num_nbs, pWna, pWU2,
                                              W_nei_bond, b_U2, fnei, nlab);
    gemm32x64_wmma<<<gemmBlocks, 128, 0, stream>>>(af16, pWself, nullptr, fself,
                                                   nullptr, MG, 10, HP, HP, HP, 0);
    if (d == DEPTH - 1) {
      wln_kernels_out<<<blk((long)MM * HH, TPB), TPB, 0, stream>>>(fnei, fself,
                                                                   node_mask, out);
    }
    wln_build_newlab<<<blk((long)MM * NLP, TPB), TPB, 0, stream>>>(af16, nlab, nl16);
    gemm32x64_wmma<<<gemmBlocks, 128, 0, stream>>>(nl16, pWU1, b_U1, af32, af16,
                                                   MG, 20, NLP, HP, HP, 1);
  }
  wln_copy_af<<<blk((long)MM * HH, TPB), TPB, 0, stream>>>(af32, out + (long)MM * HH);
}
